// CausalMultiHeadAttention_24429773980375
// MI455X (gfx1250) — matrix operand order repaired and checked by reading, not compiled, not run on hardware
//
#include <hip/hip_runtime.h>
#include <hip/hip_bf16.h>

// ---------------------------------------------------------------------------
// Causal MHA for MI455X (gfx1250, wave32).
//  - All matrix math on v_wmma_f32_16x16x32_bf16 (f32 accumulate).
//  - GEMMs: 8 waves/block share an LDS-staged bf16 W tile (128x64 macro-tile).
//  - Flash attention: 4 query-tiles/block share K/V blocks double-buffered
//    into LDS with GLOBAL_LOAD_ASYNC_TO_LDS_B128 + s_wait_asynccnt.
// ---------------------------------------------------------------------------

typedef __attribute__((ext_vector_type(16))) __bf16 v16bf;
typedef __attribute__((ext_vector_type(8)))  float  v8f;

#define B_  4
#define S_  2048
#define C_  1024
#define H_  16
#define D_  64
#define TC_ (3 * C_)   // 3072
#define GW  8          // waves per GEMM block
#define QW  4          // query tiles (waves) per attention block

union ABOp { v16bf v; unsigned u[8]; };
union Acc  { v8f   v; float    f[8]; };

__device__ __forceinline__ unsigned short f2bf(float f) {
    unsigned u = __float_as_uint(f);
    u += 0x7FFFu + ((u >> 16) & 1u);          // round-to-nearest-even
    return (unsigned short)(u >> 16);
}
__device__ __forceinline__ unsigned pack2bf(float f0, float f1) {
    return (unsigned)f2bf(f0) | ((unsigned)f2bf(f1) << 16);
}
// 16-bit A-matrix 16x32 operand order (matches B fills): lane half h, VGPR j
// holds K = {16h+2j,+1}.
__device__ __forceinline__ int a_pair_base(int h, int j) {
    return 16 * h + 2 * j;
}
__device__ __forceinline__ v8f wmma_bf16(const ABOp& a, const ABOp& b, v8f c) {
    return __builtin_amdgcn_wmma_f32_16x16x32_bf16(false, a.v, false, b.v,
                                                   (short)0, c, false, false);
}

// ---- CDNA5 async global->LDS (ASYNCcnt-tracked), inline asm -----------------
__device__ __forceinline__ void async_ld_b128(unsigned lds_off, const void* g) {
    asm volatile("global_load_async_to_lds_b128 %0, %1, off"
                 :: "v"(lds_off), "v"((unsigned long long)(size_t)g)
                 : "memory");
}
__device__ __forceinline__ void wait_async0() {
    asm volatile("s_wait_asynccnt 0x0" ::: "memory");
}
__device__ __forceinline__ void wait_async4() {
    asm volatile("s_wait_asynccnt 0x4" ::: "memory");
}
__device__ __forceinline__ unsigned lds_offset(const void* p) {
    return (unsigned)(size_t)p;   // generic addr low 32 bits == LDS offset
}

// ---------------------------------------------------------------------------
// Kernel 1: qkv = x @ W_attn + b_attn ; scatter to q[B,H,S,D], k[B,H,S,D],
// vT[B,H,D,S] bf16.  8 waves/block -> 128x64 output macro-tile.
// ---------------------------------------------------------------------------
__global__ __launch_bounds__(256)
void qkv_gemm_kernel(const float* __restrict__ x, const float* __restrict__ W,
                     const float* __restrict__ bias,
                     unsigned short* __restrict__ qws,
                     unsigned short* __restrict__ kws,
                     unsigned short* __restrict__ vtws) {
    const int tid = threadIdx.x;
    const int lane = tid & 31, wv = tid >> 5;
    const int lo = lane & 15, h = lane >> 4;
    const int row0 = blockIdx.y * (16 * GW);
    const int arow = row0 + wv * 16 + lo;
    const int col0 = blockIdx.x * 64;

    __shared__ unsigned short wtile[64 * 32];   // [n][k] bf16, shared by block

    Acc acc[4];
#pragma unroll
    for (int t = 0; t < 4; ++t)
#pragma unroll
        for (int i = 0; i < 8; ++i) acc[t].f[i] = 0.0f;

    for (int it = 0; it < C_ / 32; ++it) {
        const int kk0 = it * 32;
        __syncthreads();
        // Cooperative stage of W[kk0..+31][col0..+63] -> LDS bf16 [n][k].
#pragma unroll
        for (int i = 0; i < 8; ++i) {
            const int f = tid + 256 * i;        // 2048 elems / 256 threads
            const int k = f >> 6, n = f & 63;   // coalesced along n
            wtile[n * 32 + k] = f2bf(W[(size_t)(kk0 + k) * TC_ + col0 + n]);
        }
        __syncthreads();

        ABOp a;
#pragma unroll
        for (int j = 0; j < 8; ++j) {
            const int kb = a_pair_base(h, j);
            const float2 x2 = *reinterpret_cast<const float2*>(
                &x[(size_t)arow * C_ + kk0 + kb]);
            a.u[j] = pack2bf(x2.x, x2.y);
        }
#pragma unroll
        for (int t = 0; t < 4; ++t) {
            ABOp b;
#pragma unroll
            for (int j = 0; j < 8; ++j)   // B 32x16: lane half h -> K=16h+2j
                b.u[j] = *reinterpret_cast<const unsigned*>(
                    &wtile[(t * 16 + lo) * 32 + 16 * h + 2 * j]);
            acc[t].v = wmma_bf16(a, b, acc[t].v);
        }
    }

#pragma unroll
    for (int t = 0; t < 4; ++t) {
        const int col = col0 + t * 16 + lo;
        const float bv = bias[col];
#pragma unroll
        for (int r = 0; r < 8; ++r) {
            const int orow = row0 + wv * 16 + 8 * h + r;
            const int ob = orow >> 11, os = orow & (S_ - 1);
            const unsigned short val = f2bf(acc[t].f[r] + bv);
            if (col < C_) {
                const int hh = col >> 6, d = col & 63;
                qws[((size_t)(ob * H_ + hh) * S_ + os) * D_ + d] = val;
            } else if (col < 2 * C_) {
                const int c2 = col - C_, hh = c2 >> 6, d = c2 & 63;
                kws[((size_t)(ob * H_ + hh) * S_ + os) * D_ + d] = val;
            } else {
                const int c2 = col - 2 * C_, hh = c2 >> 6, d = c2 & 63;
                vtws[((size_t)(ob * H_ + hh) * D_ + d) * S_ + os] = val;
            }
        }
    }
}

// ---------------------------------------------------------------------------
// Kernel 2: causal flash attention.  4 waves (4 query tiles of 16 rows) per
// block share each 32-key K/V block, double-buffered into LDS with async
// global->LDS copies.  Per 32-key block: 4 WMMA (QK^T) + 4 WMMA (PV) / wave.
// ---------------------------------------------------------------------------
__global__ __launch_bounds__(32 * QW)
void flash_attn_kernel(const unsigned short* __restrict__ qws,
                       const unsigned short* __restrict__ kws,
                       const unsigned short* __restrict__ vtws,
                       unsigned short* __restrict__ aws) {
    const int tid = threadIdx.x;
    const int lane = tid & 31, w = tid >> 5;
    const int lo = lane & 15, h = lane >> 4;
    const int bh = blockIdx.y;
    const int qbase = (blockIdx.x * QW + w) * 16;          // this wave's rows
    const int qmax  = (blockIdx.x * QW + QW) * 16 - 1;     // block's last row

    const unsigned short* qp = qws + (size_t)bh * S_ * D_;
    const unsigned short* kp = kws + (size_t)bh * S_ * D_;
    const unsigned short* vp = vtws + (size_t)bh * D_ * S_;

    __shared__ unsigned short kbuf[2][32 * 64];   // [key][d] bf16, 4KB each
    __shared__ unsigned short vbuf[2][64 * 32];   // [d][key] bf16, 4KB each
    __shared__ unsigned short lds_p[QW][16 * 32]; // per-wave P tile

    // Issue one K/V block into buffer `buf` (4 async b128 per wave).
    auto stage = [&](int buf, int kb) {
#pragma unroll
        for (int i = 0; i < 2; ++i) {             // K: 4KB contiguous
            const int c = (w * 2 + i) * 32 + lane;       // 16B chunk id
            async_ld_b128(lds_offset(&kbuf[buf][0]) + c * 16,
                          (const char*)kp + (size_t)kb * 128 + c * 16);
        }
#pragma unroll
        for (int i = 0; i < 2; ++i) {             // V: 64 rows x 64B (strided)
            const int c = (w * 2 + i) * 32 + lane;
            const int row = c >> 2, off = (c & 3) * 16;
            async_ld_b128(lds_offset(&vbuf[buf][0]) + c * 16,
                          (const char*)vp + ((size_t)row * S_ + kb) * 2 + off);
        }
    };

    // Q as two A operands (d 0..31, 32..63); q is bf16 with contiguous pairs.
    ABOp qa0, qa1;
#pragma unroll
    for (int j = 0; j < 8; ++j) {
        const int kb = a_pair_base(h, j);
        qa0.u[j] = *reinterpret_cast<const unsigned*>(
            &qp[(size_t)(qbase + lo) * D_ + kb]);
        qa1.u[j] = *reinterpret_cast<const unsigned*>(
            &qp[(size_t)(qbase + lo) * D_ + 32 + kb]);
    }

    Acc o[4];
#pragma unroll
    for (int t = 0; t < 4; ++t)
#pragma unroll
        for (int i = 0; i < 8; ++i) o[t].f[i] = 0.0f;
    float mstate[8], lstate[8];
#pragma unroll
    for (int r = 0; r < 8; ++r) { mstate[r] = -1e30f; lstate[r] = 0.0f; }

    stage(0, 0);
    int cur = 0;
    for (int kbase = 0; kbase <= qmax; kbase += 32) {
        const bool havenext = (kbase + 32 <= qmax);
        if (havenext) stage(cur ^ 1, kbase + 32);        // prefetch next block
        if (havenext) wait_async4(); else wait_async0(); // current block done
        __syncthreads();                                  // all waves' chunks

        // ---- scores S[16x32] = Q @ K^T from LDS ----
        Acc s0, s1;
#pragma unroll
        for (int i = 0; i < 8; ++i) { s0.f[i] = 0.0f; s1.f[i] = 0.0f; }
        {
            ABOp bk;
#pragma unroll
            for (int j = 0; j < 8; ++j)
                bk.u[j] = *reinterpret_cast<const unsigned*>(
                    &kbuf[cur][lo * 64 + 16 * h + 2 * j]);
            s0.v = wmma_bf16(qa0, bk, s0.v);
#pragma unroll
            for (int j = 0; j < 8; ++j)
                bk.u[j] = *reinterpret_cast<const unsigned*>(
                    &kbuf[cur][lo * 64 + 32 + 16 * h + 2 * j]);
            s0.v = wmma_bf16(qa1, bk, s0.v);
#pragma unroll
            for (int j = 0; j < 8; ++j)
                bk.u[j] = *reinterpret_cast<const unsigned*>(
                    &kbuf[cur][(16 + lo) * 64 + 16 * h + 2 * j]);
            s1.v = wmma_bf16(qa0, bk, s1.v);
#pragma unroll
            for (int j = 0; j < 8; ++j)
                bk.u[j] = *reinterpret_cast<const unsigned*>(
                    &kbuf[cur][(16 + lo) * 64 + 32 + 16 * h + 2 * j]);
            s1.v = wmma_bf16(qa1, bk, s1.v);
        }

        // ---- scale + causal mask (-1e30 avoids inf-inf NaN) ----
        const int key0 = kbase + lo, key1 = kbase + 16 + lo;
#pragma unroll
        for (int r = 0; r < 8; ++r) {
            const int rowq = qbase + 8 * h + r;
            float v0 = s0.f[r] * 0.125f;                  // 1/sqrt(64)
            float v1 = s1.f[r] * 0.125f;
            if (key0 > rowq) v0 = -1e30f;
            if (key1 > rowq) v1 = -1e30f;
            s0.f[r] = v0; s1.f[r] = v1;
        }

        // ---- online softmax per row (xor-shuffle stays in 16-lane half) ----
#pragma unroll
        for (int r = 0; r < 8; ++r) {
            float rmax = fmaxf(s0.f[r], s1.f[r]);
#pragma unroll
            for (int m = 1; m <= 8; m <<= 1)
                rmax = fmaxf(rmax, __shfl_xor(rmax, m, 32));
            const float mnew = fmaxf(mstate[r], rmax);
            const float p0 = __expf(s0.f[r] - mnew);
            const float p1 = __expf(s1.f[r] - mnew);
            lds_p[w][(8 * h + r) * 32 + lo]      = f2bf(p0);
            lds_p[w][(8 * h + r) * 32 + 16 + lo] = f2bf(p1);
            float rsum = p0 + p1;
#pragma unroll
            for (int m = 1; m <= 8; m <<= 1)
                rsum += __shfl_xor(rsum, m, 32);
            const float alpha = __expf(mstate[r] - mnew);
            lstate[r] = lstate[r] * alpha + rsum;
            mstate[r] = mnew;
#pragma unroll
            for (int t = 0; t < 4; ++t) o[t].f[r] *= alpha;
        }
        // same-wave DS ops are in-order; just stop compiler reordering
        asm volatile("" ::: "memory");

        // ---- repack P into A layout ----
        ABOp pa;
#pragma unroll
        for (int j = 0; j < 8; ++j) {
            const int kb = a_pair_base(h, j);
            pa.u[j] = *reinterpret_cast<const unsigned*>(
                &lds_p[w][lo * 32 + kb]);
        }

        // ---- O[16x64] += P[16x32] @ V[32x64] ----
#pragma unroll
        for (int t = 0; t < 4; ++t) {
            ABOp bv;
#pragma unroll
            for (int j = 0; j < 8; ++j)
                bv.u[j] = *reinterpret_cast<const unsigned*>(
                    &vbuf[cur][(t * 16 + lo) * 32 + 16 * h + 2 * j]);
            o[t].v = wmma_bf16(pa, bv, o[t].v);
        }
        __syncthreads();          // all reads of buf done before its reuse
        cur ^= 1;
    }

    // ---- normalize and store to attn workspace [B,S,C] bf16 ----
    const int b = bh >> 4, head = bh & 15;
#pragma unroll
    for (int r = 0; r < 8; ++r) {
        const float inv = (lstate[r] > 0.0f) ? (1.0f / lstate[r]) : 0.0f;
        const int srow = qbase + 8 * h + r;
#pragma unroll
        for (int t = 0; t < 4; ++t)
            aws[(size_t)(b * S_ + srow) * C_ + head * 64 + t * 16 + lo] =
                f2bf(o[t].f[r] * inv);
    }
}

// ---------------------------------------------------------------------------
// Kernel 3: out = attn @ W_proj + b_proj (bf16 A from ws, f32 out).
// ---------------------------------------------------------------------------
__global__ __launch_bounds__(256)
void proj_gemm_kernel(const unsigned short* __restrict__ aws,
                      const float* __restrict__ W,
                      const float* __restrict__ bias,
                      float* __restrict__ out) {
    const int tid = threadIdx.x;
    const int lane = tid & 31, wv = tid >> 5;
    const int lo = lane & 15, h = lane >> 4;
    const int row0 = blockIdx.y * (16 * GW);
    const int arow = row0 + wv * 16 + lo;
    const int col0 = blockIdx.x * 64;

    __shared__ unsigned short wtile[64 * 32];

    Acc acc[4];
#pragma unroll
    for (int t = 0; t < 4; ++t)
#pragma unroll
        for (int i = 0; i < 8; ++i) acc[t].f[i] = 0.0f;

    for (int it = 0; it < C_ / 32; ++it) {
        const int kk0 = it * 32;
        __syncthreads();
#pragma unroll
        for (int i = 0; i < 8; ++i) {
            const int f = tid + 256 * i;
            const int k = f >> 6, n = f & 63;
            wtile[n * 32 + k] = f2bf(W[(size_t)(kk0 + k) * C_ + col0 + n]);
        }
        __syncthreads();

        ABOp a;
#pragma unroll
        for (int j = 0; j < 8; ++j) {
            const int kb = a_pair_base(h, j);
            a.u[j] = *reinterpret_cast<const unsigned*>(
                &aws[(size_t)arow * C_ + kk0 + kb]);
        }
#pragma unroll
        for (int t = 0; t < 4; ++t) {
            ABOp b;
#pragma unroll
            for (int j = 0; j < 8; ++j)
                b.u[j] = *reinterpret_cast<const unsigned*>(
                    &wtile[(t * 16 + lo) * 32 + 16 * h + 2 * j]);
            acc[t].v = wmma_bf16(a, b, acc[t].v);
        }
    }

#pragma unroll
    for (int t = 0; t < 4; ++t) {
        const int col = col0 + t * 16 + lo;
        const float bv = bias[col];
#pragma unroll
        for (int r = 0; r < 8; ++r) {
            const int orow = row0 + wv * 16 + 8 * h + r;
            out[(size_t)orow * C_ + col] = acc[t].f[r] + bv;
        }
    }
}

// ---------------------------------------------------------------------------
extern "C" void kernel_launch(void* const* d_in, const int* in_sizes, int n_in,
                              void* d_out, int out_size, void* d_ws,
                              size_t ws_size, hipStream_t stream) {
    (void)in_sizes; (void)n_in; (void)out_size; (void)ws_size;
    const float* x      = (const float*)d_in[0];
    const float* W_attn = (const float*)d_in[1];
    const float* b_attn = (const float*)d_in[2];
    const float* W_proj = (const float*)d_in[3];
    const float* b_proj = (const float*)d_in[4];
    float* out = (float*)d_out;

    const size_t per = (size_t)B_ * H_ * S_ * D_;       // 8.4M elems / region
    unsigned short* qws  = (unsigned short*)d_ws;
    unsigned short* kws  = qws + per;
    unsigned short* vtws = kws + per;
    unsigned short* aws  = vtws + per;

    qkv_gemm_kernel<<<dim3(TC_ / 64, (B_ * S_) / (16 * GW)), 256, 0, stream>>>(
        x, W_attn, b_attn, qws, kws, vtws);
    flash_attn_kernel<<<dim3(S_ / 16 / QW, B_ * H_), 32 * QW, 0, stream>>>(
        qws, kws, vtws, aws);
    proj_gemm_kernel<<<dim3(C_ / 64, (B_ * S_) / (16 * GW)), 256, 0, stream>>>(
        aws, W_proj, b_proj, out);
}
